// Kmodel_13855564497544
// MI455X (gfx1250) — matrix operand order repaired and checked by reading, not compiled, not run on hardware
//
#include <hip/hip_runtime.h>

// ---------------------------------------------------------------------------
// k-means assignment + update for MI455X (gfx1250, wave32, WMMA)
//   d2 = |x|^2 + |c|^2 - 2 x.c  via split-bf16 (hi/lo) WMMA GEMM, f32 accum
//   argmin + loss + segment scatter-sum with global f32 atomics
//
// Round-2 change: 3 independent WMMA accumulator chains (one per split
// product) instead of one serial 48-deep D->C chain, so the matrix pipe can
// interleave ops without s_delay_alu stalls.
// ---------------------------------------------------------------------------

typedef __attribute__((ext_vector_type(16))) __bf16 v16bf;
typedef __attribute__((ext_vector_type(8)))  float  v8f;

#define NN 65536
#define DD 512
#define KK 2048

// round-to-nearest-even f32 -> bf16 split: f ~= hi + lo (both bf16)
__device__ __forceinline__ void bf16_split(float f, unsigned short& h, unsigned short& l) {
    unsigned u  = __float_as_uint(f);
    unsigned rh = u + 0x7FFFu + ((u >> 16) & 1u);
    h = (unsigned short)(rh >> 16);
    float fh = __uint_as_float(rh & 0xFFFF0000u);
    float r  = f - fh;
    unsigned v  = __float_as_uint(r);
    unsigned rl = v + 0x7FFFu + ((v >> 16) & 1u);
    l = (unsigned short)(rl >> 16);
}

// ---------------------------------------------------------------------------
// Zero the accumulated output regions (loss, new_cluster, counts)
// ---------------------------------------------------------------------------
__global__ void zero_tail(float* __restrict__ p, long n) {
    long i = (long)blockIdx.x * blockDim.x + threadIdx.x;
    long s = (long)gridDim.x * blockDim.x;
    for (; i < n; i += s) p[i] = 0.0f;
}

// ---------------------------------------------------------------------------
// cluster [K,D] f32  ->  ch/cl bf16 [K,D] + c2 [K]   (one wave per row)
// ---------------------------------------------------------------------------
__global__ void prep_clusters(const float* __restrict__ cluster,
                              unsigned short* __restrict__ ch,
                              unsigned short* __restrict__ cl,
                              float* __restrict__ c2) {
    int row  = (blockIdx.x * blockDim.x + threadIdx.x) >> 5;
    int lane = threadIdx.x & 31;
    if (row >= KK) return;
    const float* src = cluster + (size_t)row * DD;
    float ss = 0.0f;
    for (int e = lane; e < DD; e += 32) {
        float f = src[e];
        ss += f * f;
        unsigned short h, l;
        bf16_split(f, h, l);
        ch[(size_t)row * DD + e] = h;
        cl[(size_t)row * DD + e] = l;
    }
    #pragma unroll
    for (int m = 16; m; m >>= 1) ss += __shfl_xor(ss, m, 32);
    if (lane == 0) c2[row] = ss;
}

// ---------------------------------------------------------------------------
// Main kernel: 256 threads = 8 waves, each wave owns a 16-row strip.
// A fragments (x, split bf16) resident in VGPRs; clusters streamed from L2.
// ---------------------------------------------------------------------------
__global__ __launch_bounds__(256) void kmeans_main(
        const float*          __restrict__ x,
        const unsigned short* __restrict__ ch,
        const unsigned short* __restrict__ cl,
        const float*          __restrict__ c2,
        int*                  __restrict__ idx_out,
        float*                __restrict__ loss_out,
        float*                __restrict__ newc,   // [K,D]
        float*                __restrict__ newn)   // [K]
{
    const int tid     = threadIdx.x;
    const int wave    = tid >> 5;
    const int lane    = tid & 31;
    const int m       = lane & 15;      // row (A) / column (B) within tile
    const int hi      = lane >> 4;      // half-wave selector
    const int rowbase = blockIdx.x * 128 + wave * 16;

    // ---- Load & split x rows into A fragments (contiguous k = 16*hi + i order, same as B) ----
    // lane l (m = l&15, hi = l>>4) holds, for chunk kc (D range 32*kc..+31):
    //   a[0..15] -> d = 32*kc + hi*16 + (0..15)
    v16bf Ah[16], Al[16];
    float ss = 0.0f;
    const float* xrow = x + (size_t)(rowbase + m) * DD;
    #pragma unroll
    for (int kc = 0; kc < 16; ++kc) {
        const float* p = xrow + kc * 32 + hi * 16;
        float f[16];
        float4 a0 = *(const float4*)(p + 0);
        float4 a1 = *(const float4*)(p + 4);
        float4 a2 = *(const float4*)(p + 8);
        float4 a3 = *(const float4*)(p + 12);
        f[0]=a0.x;  f[1]=a0.y;  f[2]=a0.z;  f[3]=a0.w;
        f[4]=a1.x;  f[5]=a1.y;  f[6]=a1.z;  f[7]=a1.w;
        f[8]=a2.x;  f[9]=a2.y;  f[10]=a2.z; f[11]=a2.w;
        f[12]=a3.x; f[13]=a3.y; f[14]=a3.z; f[15]=a3.w;
        union { v16bf v; unsigned short u[16]; } uh, ul;
        #pragma unroll
        for (int i = 0; i < 16; ++i) {
            ss += f[i] * f[i];
            unsigned short h, l;
            bf16_split(f[i], h, l);
            uh.u[i] = h;
            ul.u[i] = l;
        }
        Ah[kc] = uh.v;
        Al[kc] = ul.v;
    }

    // |x|^2 per row: lane l and l^16 hold complementary halves of row m
    float x2own = ss + __shfl_xor(ss, 16, 32);
    // C/D layout: (VGPR r, lane l) -> M = r + 8*hi, N = l&15
    float x2r[8];
    #pragma unroll
    for (int r = 0; r < 8; ++r) x2r[r] = __shfl(x2own, r + hi * 8, 32);

    float best[8];
    int   bidx[8];
    #pragma unroll
    for (int r = 0; r < 8; ++r) { best[r] = 3.4e38f; bidx[r] = 0; }

    // ---- Sweep all K in 16-column tiles ----
    for (int kt = 0; kt < KK / 16; ++kt) {
        const int   col = kt * 16 + m;                 // this lane's B column
        const float c2v = c2[col];
        // B layout (32x16, 16-bit): lane l holds column n=l&15, K = 16*hi + i
        const unsigned short* bhp = ch + (size_t)col * DD + hi * 16;
        const unsigned short* blp = cl + (size_t)col * DD + hi * 16;

        // Three independent accumulator chains: xh*ch, xh*cl, xl*ch
        v8f acc0 = {0.f, 0.f, 0.f, 0.f, 0.f, 0.f, 0.f, 0.f};
        v8f acc1 = {0.f, 0.f, 0.f, 0.f, 0.f, 0.f, 0.f, 0.f};
        v8f acc2 = {0.f, 0.f, 0.f, 0.f, 0.f, 0.f, 0.f, 0.f};
        #pragma unroll
        for (int kc = 0; kc < 16; ++kc) {
            v16bf Bh = *(const v16bf*)(bhp + kc * 32);
            v16bf Bl = *(const v16bf*)(blp + kc * 32);
            acc0 = __builtin_amdgcn_wmma_f32_16x16x32_bf16(false, Ah[kc], false, Bh,
                                                           (short)0, acc0, false, false);
            acc1 = __builtin_amdgcn_wmma_f32_16x16x32_bf16(false, Ah[kc], false, Bl,
                                                           (short)0, acc1, false, false);
            acc2 = __builtin_amdgcn_wmma_f32_16x16x32_bf16(false, Al[kc], false, Bh,
                                                           (short)0, acc2, false, false);
        }
        #pragma unroll
        for (int r = 0; r < 8; ++r) {
            float dot = acc0[r] + acc1[r] + acc2[r];
            float d2  = __builtin_fmaf(-2.0f, dot, x2r[r] + c2v);
            if (d2 < best[r]) { best[r] = d2; bidx[r] = col; }
        }
    }

    // ---- argmin reduce across the 16 lanes sharing each row (first-index ties) ----
    #pragma unroll
    for (int r = 0; r < 8; ++r) {
        #pragma unroll
        for (int off = 1; off < 16; off <<= 1) {
            float ob = __shfl_xor(best[r], off, 32);
            int   oi = __shfl_xor(bidx[r], off, 32);
            if (ob < best[r] || (ob == best[r] && oi < bidx[r])) {
                best[r] = ob; bidx[r] = oi;
            }
        }
    }

    if (m == 0) {   // lanes 0 (rows 0..7) and 16 (rows 8..15)
        float lsum = 0.0f;
        #pragma unroll
        for (int r = 0; r < 8; ++r) {
            idx_out[rowbase + hi * 8 + r] = bidx[r];
            lsum += best[r];
        }
        atomicAdd(loss_out, lsum * (1.0f / (float)NN));
    }

    // ---- segment scatter-sum: each wave accumulates its own 16 rows ----
    #pragma unroll
    for (int s = 0; s < 16; ++s) {
        int c = __shfl(bidx[s & 7], (s >= 8) ? 16 : 0, 32);
        const float* xr  = x + (size_t)(rowbase + s) * DD;
        float*       dst = newc + (size_t)c * DD;
        #pragma unroll 4
        for (int e = lane; e < DD; e += 32) atomicAdd(dst + e, xr[e]);
        if (lane == 0) atomicAdd(newn + c, 1.0f);
    }
}

// ---------------------------------------------------------------------------
extern "C" void kernel_launch(void* const* d_in, const int* in_sizes, int n_in,
                              void* d_out, int out_size, void* d_ws, size_t ws_size,
                              hipStream_t stream) {
    (void)in_sizes; (void)n_in; (void)out_size; (void)ws_size;

    const float* x       = (const float*)d_in[0];   // [N,D]
    const float* cluster = (const float*)d_in[1];   // [K,D]

    // Output layout (reference return order, concatenated flat):
    //   idx[N] (int32), loss[1], new_cluster[K*D], new_cluster_num[K]
    int*   idx_out = (int*)d_out;
    float* loss    = (float*)d_out + NN;
    float* newc    = loss + 1;
    float* newn    = newc + (size_t)KK * DD;

    // Workspace: ch[K*D] bf16, cl[K*D] bf16, c2[K] f32  (~4.20 MB)
    unsigned short* ch = (unsigned short*)d_ws;
    unsigned short* cl = ch + (size_t)KK * DD;
    float*          c2 = (float*)(cl + (size_t)KK * DD);

    const long tail = 1L + (long)KK * DD + KK;
    zero_tail<<<2056, 256, 0, stream>>>(loss, tail);
    prep_clusters<<<KK / 8, 256, 0, stream>>>(cluster, ch, cl, c2);
    kmeans_main<<<NN / 128, 256, 0, stream>>>(x, ch, cl, c2, idx_out, loss, newc, newn);
}
